// Interaction_GraphConvolution_60919816126908
// MI455X (gfx1250) — compile-verified, matrix operand order checked by reading, not run on hardware
//
#include <hip/hip_runtime.h>
#include <hip/hip_bf16.h>

typedef __attribute__((ext_vector_type(16))) _Float16 v16h;
typedef __attribute__((ext_vector_type(8)))  float    v8f;

#define BM 128
#define BN 128
#define BK 32
#define LDS_STRIDE (BK + 8)   // 40 halves -> 80B row stride (multiple of 16B)

#if defined(__gfx1250__) && \
    __has_builtin(__builtin_amdgcn_global_load_async_to_lds_b128)
#define USE_ASYNC_LDS 1
#else
#define USE_ASYNC_LDS 0
#endif

union Frag16 { uint4 u[2]; v16h v; };

enum { EPI_X16_BIAS = 1, EPI_WF = 2, EPI_P = 3, EPI_OUT = 4 };

#if USE_ASYNC_LDS
typedef __attribute__((__vector_size__(4 * sizeof(int)))) int i32x4;
typedef i32x4 __attribute__((address_space(1))) gi32x4;  // global
typedef i32x4 __attribute__((address_space(3))) li32x4;  // LDS
#endif

__device__ __forceinline__ void async_cp16(const _Float16* g, _Float16* l) {
#if USE_ASYNC_LDS
  __builtin_amdgcn_global_load_async_to_lds_b128((gi32x4*)g, (li32x4*)l, 0, 0);
#else
  (void)g; (void)l;
#endif
}

__device__ __forceinline__ void wait_async0() {
#if defined(__gfx1250__) && __has_builtin(__builtin_amdgcn_s_wait_asynccnt)
  __builtin_amdgcn_s_wait_asynccnt(0);
#elif defined(__gfx1250__)
  asm volatile("s_wait_asynccnt 0" ::: "memory");
#endif
}

// C[m,n] = sum_k A[m,k] * Bt[n,k]   (A: MxK row-major f16, Bt: NxK row-major f16)
// 256 threads = 8 wave32s; block tile 128x128x32; wave tile 32x64 (2x4 WMMA frags)
// Double-buffered LDS, one barrier per k-step; staging overlaps compute.
template <int EPI>
__global__ __launch_bounds__(256) void gemm_wmma_nt(
    const _Float16* __restrict__ A,
    const _Float16* __restrict__ Bt,
    int M, int N, int K,
    float*    __restrict__ Cf,     // f32 output (EPI_WF: WF, EPI_OUT: final out)
    _Float16* __restrict__ Ch,     // f16 output (X16 / WFt16 / P16)
    const float* __restrict__ ep0, // bias / S / WF
    const float* __restrict__ ep1) // neighbor_count
{
  __shared__ __align__(16) _Float16 As[2][BM][LDS_STRIDE];
  __shared__ __align__(16) _Float16 Bs[2][BN][LDS_STRIDE];

  const int t    = threadIdx.x;
  const int lane = t & 31;
  const int wave = t >> 5;
  const int wm   = (wave >> 1) * 32;   // 4 wave-rows along M
  const int wn   = (wave & 1) * 64;    // 2 wave-cols along N
  const int bm   = blockIdx.y * BM;
  const int bn   = blockIdx.x * BN;
  const int hi   = (lane < 16) ? 0 : 8; // K-half select per WMMA 16-bit A/B layout
  const int l16  = lane & 15;

  // Staging coords: tile is 128 rows x 32 halves = 512 uint4; thread t owns
  // rows sr and sr+64 at half-column sc (one b128 each).
  const int sr = t >> 2;            // 0..63
  const int sc = (t & 3) << 3;      // 0,8,16,24
  const _Float16* ga = A  + (size_t)(bm + sr) * K + sc;
  const _Float16* gb = Bt + (size_t)(bn + sr) * K + sc;
  const size_t rowskip = (size_t)64 * K;   // +64 rows

  const v8f vzero = {0.f, 0.f, 0.f, 0.f, 0.f, 0.f, 0.f, 0.f};
  v8f acc[2][4];
#pragma unroll
  for (int i = 0; i < 2; ++i)
#pragma unroll
    for (int j = 0; j < 4; ++j) acc[i][j] = vzero;

  uint4 ra0, ra1, rb0, rb1;  // register staging (fallback path)

  auto stage_async = [&](int k0, int b) {
    async_cp16(ga + k0,           &As[b][sr][sc]);
    async_cp16(ga + k0 + rowskip, &As[b][sr + 64][sc]);
    async_cp16(gb + k0,           &Bs[b][sr][sc]);
    async_cp16(gb + k0 + rowskip, &Bs[b][sr + 64][sc]);
  };
  auto stage_load = [&](int k0) {
    ra0 = *reinterpret_cast<const uint4*>(ga + k0);
    ra1 = *reinterpret_cast<const uint4*>(ga + k0 + rowskip);
    rb0 = *reinterpret_cast<const uint4*>(gb + k0);
    rb1 = *reinterpret_cast<const uint4*>(gb + k0 + rowskip);
  };
  auto stage_store = [&](int b) {
    *reinterpret_cast<uint4*>(&As[b][sr][sc])      = ra0;
    *reinterpret_cast<uint4*>(&As[b][sr + 64][sc]) = ra1;
    *reinterpret_cast<uint4*>(&Bs[b][sr][sc])      = rb0;
    *reinterpret_cast<uint4*>(&Bs[b][sr + 64][sc]) = rb1;
  };
  auto compute = [&](int b) {
    Frag16 fa[2], fb[4];
#pragma unroll
    for (int tm = 0; tm < 2; ++tm) {
      const int row = wm + tm * 16 + l16;
      fa[tm].u[0] = *reinterpret_cast<const uint4*>(&As[b][row][hi]);
      fa[tm].u[1] = *reinterpret_cast<const uint4*>(&As[b][row][hi + 16]);
    }
#pragma unroll
    for (int tn = 0; tn < 4; ++tn) {
      const int row = wn + tn * 16 + l16;
      fb[tn].u[0] = *reinterpret_cast<const uint4*>(&Bs[b][row][hi]);
      fb[tn].u[1] = *reinterpret_cast<const uint4*>(&Bs[b][row][hi + 16]);
    }
#pragma unroll
    for (int tm = 0; tm < 2; ++tm)
#pragma unroll
      for (int tn = 0; tn < 4; ++tn)
        acc[tm][tn] = __builtin_amdgcn_wmma_f32_16x16x32_f16(
            false, fa[tm].v, false, fb[tn].v, (short)0, acc[tm][tn], false, false);
  };

  int buf = 0;
#if USE_ASYNC_LDS
  stage_async(0, 0);
  for (int k0 = 0; k0 < K; k0 += BK) {
    wait_async0();          // this wave's LDS tile writes landed
    __syncthreads();        // everyone's tile visible
    if (k0 + BK < K) stage_async(k0 + BK, buf ^ 1);  // overlap with compute
    compute(buf);
    buf ^= 1;
  }
#else
  stage_load(0);
  for (int k0 = 0; k0 < K; k0 += BK) {
    stage_store(buf);       // compiler waits loadcnt here
    __syncthreads();
    if (k0 + BK < K) stage_load(k0 + BK);  // loads overlap compute
    compute(buf);
    buf ^= 1;
  }
#endif

  // Epilogue. D layout: lane holds col n = lane%16; VGPR r holds row (lane<16?0:8)+r.
#pragma unroll
  for (int tm = 0; tm < 2; ++tm) {
#pragma unroll
    for (int tn = 0; tn < 4; ++tn) {
      const int mbase = bm + wm + tm * 16 + hi;
      const int n     = bn + wn + tn * 16 + l16;
#pragma unroll
      for (int r = 0; r < 8; ++r) {
        const int m   = mbase + r;
        const float c = acc[tm][tn][r];
        if (EPI == EPI_X16_BIAS) {
          Ch[(size_t)m * N + n] = (_Float16)(c + ep0[n]);
        } else if (EPI == EPI_WF) {
          Cf[(size_t)m * N + n] = c;                 // WF (f32) for final epilogue
          Ch[(size_t)n * M + m] = (_Float16)c;       // WF^T (f16) for GEMM4 B-operand
        } else if (EPI == EPI_P) {
          // P[m,n] = G[m,n] * S[n,m]
          Ch[(size_t)m * N + n] = (_Float16)(c * ep0[(size_t)n * M + m]);
        } else if (EPI == EPI_OUT) {
          // out[n, m] = WF[m,n] * PWF[m,n] / nc[n]^2
          const float ncv = ep1[n];
          Cf[(size_t)n * M + m] = c * ep0[(size_t)m * N + n] / (ncv * ncv);
        }
      }
    }
  }
}

__global__ void k_cvt_f16(const float* __restrict__ s, _Float16* __restrict__ d,
                          size_t n) {
  size_t i = (size_t)blockIdx.x * blockDim.x + threadIdx.x;
  if (i < n) d[i] = (_Float16)s[i];
}

// dst[c*R + r] = (f16) src[r*C + c]   (R, C multiples of 32)
__global__ void k_cvt_t_f16(const float* __restrict__ s, _Float16* __restrict__ d,
                            int R, int C) {
  __shared__ float tile[32][33];
  const int tx = threadIdx.x & 31;
  const int ty = threadIdx.x >> 5;  // 0..7
  const int c0 = blockIdx.x * 32;
  const int r0 = blockIdx.y * 32;
  for (int i = 0; i < 32; i += 8)
    tile[ty + i][tx] = s[(size_t)(r0 + ty + i) * C + (c0 + tx)];
  __syncthreads();
  for (int i = 0; i < 32; i += 8)
    d[(size_t)(c0 + ty + i) * R + (r0 + tx)] = (_Float16)tile[tx][ty + i];
}

extern "C" void kernel_launch(void* const* d_in, const int* in_sizes, int n_in,
                              void* d_out, int out_size, void* d_ws, size_t ws_size,
                              hipStream_t stream) {
  const float* node_features = (const float*)d_in[0];  // 4096 x 512
  const float* adjacency     = (const float*)d_in[1];  // 4096 x 4096
  /* d_in[2] = mask_father == adjacency (unused directly) */
  const float* nc            = (const float*)d_in[3];  // 4096 x 1
  const float* S             = (const float*)d_in[4];  // 4096 x 1 x 4096 (flat NxN)
  const float* linear_w      = (const float*)d_in[5];  // 1024 x 512
  const float* linear_b      = (const float*)d_in[6];  // 1024
  const float* weight        = (const float*)d_in[7];  // 1024 x 4096

  const int Nn = 4096, FR = 512, FI = 1024, FO = 4096;

  char* w = (char*)d_ws;
  _Float16* NF16  = (_Float16*)w; w += (size_t)Nn * FR * 2;  // node_features f16
  _Float16* LW16  = (_Float16*)w; w += (size_t)FI * FR * 2;  // linear_w (already NxK)
  _Float16* X16   = (_Float16*)w; w += (size_t)Nn * FI * 2;  // X f16
  _Float16* Wt16  = (_Float16*)w; w += (size_t)FO * FI * 2;  // weight^T f16
  _Float16* AT16  = (_Float16*)w; w += (size_t)Nn * Nn * 2;  // adjacency^T f16
  _Float16* P16   = (_Float16*)w; w += (size_t)Nn * Nn * 2;  // P f16
  _Float16* WFt16 = (_Float16*)w; w += (size_t)Nn * FO * 2;  // WF^T f16
  float*    WF    = (float*)w;    w += (size_t)Nn * FO * 4;  // WF f32

  // Precision staging (f16 is exact for adjacency 0/1 and integer counts in G/S).
  k_cvt_f16<<<(Nn * FR) / 256, 256, 0, stream>>>(node_features, NF16, (size_t)Nn * FR);
  k_cvt_f16<<<(FI * FR) / 256, 256, 0, stream>>>(linear_w, LW16, (size_t)FI * FR);
  k_cvt_t_f16<<<dim3(FO / 32, FI / 32), 256, 0, stream>>>(weight, Wt16, FI, FO);
  k_cvt_t_f16<<<dim3(Nn / 32, Nn / 32), 256, 0, stream>>>(adjacency, AT16, Nn, Nn);

  // GEMM1: X = NF @ LW^T + b          (M=4096, N=1024, K=512)
  gemm_wmma_nt<EPI_X16_BIAS><<<dim3(FI / BN, Nn / BM), 256, 0, stream>>>(
      NF16, LW16, Nn, FI, FR, nullptr, X16, linear_b, nullptr);

  // GEMM2: WF = X @ weight            (M=4096, N=4096, K=1024); also emits WF^T f16
  gemm_wmma_nt<EPI_WF><<<dim3(FO / BN, Nn / BM), 256, 0, stream>>>(
      X16, Wt16, Nn, FO, FI, WF, WFt16, nullptr, nullptr);

  // GEMM3: P = (A^T @ A) .* S^T       (M=N=K=4096); both operands are A^T in NT form
  gemm_wmma_nt<EPI_P><<<dim3(Nn / BN, Nn / BM), 256, 0, stream>>>(
      AT16, AT16, Nn, Nn, Nn, nullptr, P16, S, nullptr);

  // GEMM4: out = (WF .* (P @ WF))^T / nc^2   (M=4096, N=4096, K=4096), fused final
  gemm_wmma_nt<EPI_OUT><<<dim3(FO / BN, Nn / BM), 256, 0, stream>>>(
      P16, WFt16, Nn, FO, Nn, (float*)d_out, nullptr, WF, nc);
}
